// GraphFeatLearningLayer_41480794145238
// MI455X (gfx1250) — matrix operand order repaired and checked by reading, not compiled, not run on hardware
//
#include <hip/hip_runtime.h>
#include <hip/hip_bf16.h>

// GraphFeatLearning diffusion features on MI455X (gfx1250, wave32, WMMA).
// G = 16 graphs, N = 2048, d = 128.
//   y = 0.5 * Kraw @ (x / colsum) + 0.5 * x      (P never materialized)
// Kraw kept in f16 (134 MB -> L2-resident across the 8 steps).
// Diffusion GEMM uses async global->LDS staging (ASYNCcnt) double-buffered
// against v_wmma_f32_16x16x32_f16.

typedef __attribute__((ext_vector_type(16))) _Float16 v16h;
typedef __attribute__((ext_vector_type(8)))  _Float16 v8h;
typedef __attribute__((ext_vector_type(8)))  float    v8f;
typedef __attribute__((ext_vector_type(4)))  int      v4i;

#define NG 16      // graphs (B * n_w)
#define NP 2048    // points per graph
#define ND 128     // feature dims

#define AS1 __attribute__((address_space(1)))
#define AS3 __attribute__((address_space(3)))

#if __has_builtin(__builtin_amdgcn_global_load_async_to_lds_b128)
#define HAVE_ASYNC_LDS 1
#else
#define HAVE_ASYNC_LDS 0
#endif

// 16-byte global -> LDS copy (async DMA when available).
// Builtin signature (from hipcc diagnostic): (v4i __device__*, v4i __shared__*,
// imm offset, imm cpol).
__device__ __forceinline__ void copy16_to_lds(const void* g, void* l) {
#if HAVE_ASYNC_LDS
  __builtin_amdgcn_global_load_async_to_lds_b128(
      (AS1 v4i*)(void*)g, (AS3 v4i*)l, 0, 0);
#else
  *(v8h*)l = *(const v8h*)g;
#endif
}

#if HAVE_ASYNC_LDS
#if __has_builtin(__builtin_amdgcn_s_wait_asynccnt)
#define WAIT_ASYNC(n) __builtin_amdgcn_s_wait_asynccnt(n)
#else
#define WAIT_ASYNC(n) asm volatile("s_wait_asynccnt %0" ::"i"(n) : "memory")
#endif
#else
#define WAIT_ASYNC(n) ((void)0)
#endif

// ---- WMMA fragment helpers (ISA 7.12.2 16-bit layouts, wave32) -------------
__device__ __forceinline__ v16h join16(v8h lo, v8h hi) {
  v16h a;
#pragma unroll
  for (int i = 0; i < 8; ++i) { a[i] = lo[i]; a[i + 8] = hi[i]; }
  return a;
}
// A (16x32): lane m = lane&15, h = lane>>4:
//   a[0..7]=row[kb+h*8 ..], a[8..15]=row[kb+16+h*8 ..]
__device__ __forceinline__ v16h load_a_frag(const _Float16* __restrict__ row,
                                            int kb, int h) {
  return join16(*(const v8h*)(row + kb + h * 8),
                *(const v8h*)(row + kb + 16 + h * 8));
}
// B (32x16): lane n = lane&15, h = lane>>4:
//   b[0..7]=Brow_n[kb+h*8 ..], b[8..15]=Brow_n[kb+16+h*8 ..]
__device__ __forceinline__ v16h load_b_frag(const _Float16* __restrict__ row,
                                            int kb, int h) {
  return join16(*(const v8h*)(row + kb + h * 8),
                *(const v8h*)(row + kb + 16 + h * 8));
}
__device__ __forceinline__ v8f wmma_f16(v16h a, v16h b, v8f c) {
  return __builtin_amdgcn_wmma_f32_16x16x32_f16(false, a, false, b,
                                                (short)0, c, false, false);
}

// ---- 1) Xb = pc*alpha (f32 + f16), row squared norms, zero colsum ----------
__global__ void xbar_kernel(const float* __restrict__ pc,
                            const float* __restrict__ al,
                            _Float16* __restrict__ Xh,
                            float* __restrict__ x0,
                            float* __restrict__ rn,
                            float* __restrict__ colsum) {
  const int g = blockIdx.y, n = blockIdx.x, d = threadIdx.x;
  const int b = g >> 2, w = g & 3;
  const float v = pc[((size_t)b * NP + n) * ND + d] * al[w * ND + d];
  const size_t idx = ((size_t)g * NP + n) * ND + d;
  x0[idx] = v;
  Xh[idx] = (_Float16)v;
  __shared__ float sm[ND];
  sm[d] = v * v;
  __syncthreads();
#pragma unroll
  for (int s = ND / 2; s > 0; s >>= 1) {
    if (d < s) sm[d] += sm[d + s];
    __syncthreads();
  }
  if (d == 0) {
    rn[g * NP + n]     = sm[0];
    colsum[g * NP + n] = 0.0f;
  }
}

// ---- 2) Gram via WMMA -> D -> exp/threshold -> Kraw(f16) + colsum ----------
// 8 waves/block; wave computes a 16(M) x 64(N) strip; K loop = 128 dims.
__global__ void __launch_bounds__(256)
gram_kernel(const _Float16* __restrict__ Xh, const float* __restrict__ rn,
            float* __restrict__ colsum, _Float16* __restrict__ Kh,
            const float* __restrict__ sigma_ptr) {
  const int g    = blockIdx.z;
  const int lane = threadIdx.x & 31;
  const int wave = threadIdx.x >> 5;
  const int i0   = blockIdx.y * 128 + wave * 16;  // M strip (rows)
  const int j0   = blockIdx.x * 64;               // N strip (cols)
  const int m = lane & 15, h = lane >> 4;
  const float invSigma = 1.0f / sigma_ptr[0];

  const _Float16* Xg   = Xh + (size_t)g * NP * ND;
  const _Float16* aRow = Xg + (size_t)(i0 + m) * ND;

  v8f acc[4] = {};
#pragma unroll
  for (int kb = 0; kb < ND; kb += 32) {
    const v16h a = load_a_frag(aRow, kb, h);
#pragma unroll
    for (int nt = 0; nt < 4; ++nt) {
      const _Float16* bRow = Xg + (size_t)(j0 + nt * 16 + m) * ND;
      acc[nt] = wmma_f16(a, load_b_frag(bRow, kb, h), acc[nt]);
    }
  }

  const float THR = 1e-5f;
  const float* rng = rn + (size_t)g * NP;
  _Float16* Kg = Kh + (size_t)g * NP * NP;
#pragma unroll
  for (int nt = 0; nt < 4; ++nt) {
    const int cI = j0 + nt * 16 + m;
    const float rc = rng[cI];
    float s = 0.0f;
#pragma unroll
    for (int r = 0; r < 8; ++r) {
      const int rI = i0 + 8 * h + r;
      const float Dd = rng[rI] + rc - 2.0f * acc[nt][r];
      float Kv = __expf(-Dd * invSigma);
      Kv = (Kv < THR) ? 0.0f : Kv;
      Kg[(size_t)rI * NP + cI] = (_Float16)Kv;
      s += Kv;
    }
    atomicAdd(&colsum[g * NP + cI], s);  // colsum[c] = sum over rows
  }
}

// ---- 3) z_t[g][d][c] = (f16)(x[g][c][d] / colsum[g][c])  (transposed) ------
__global__ void prescale_kernel(const float* __restrict__ x,
                                const float* __restrict__ colsum,
                                _Float16* __restrict__ zt) {
  const int g = blockIdx.y, c = blockIdx.x, d = threadIdx.x;
  const float inv = 1.0f / colsum[g * NP + c];
  const float v = x[((size_t)g * NP + c) * ND + d] * inv;
  zt[((size_t)g * ND + d) * NP + c] = (_Float16)v;
}

// ---- 4) y = 0.5 * Kraw @ z + 0.5 * x  (M=2048, N=128, K=2048) --------------
// 4 waves/block, M strip 64, full N=128. A (K rows) and B (zt, shared by all
// waves) are async-staged into LDS, double-buffered against the WMMAs.
#define MTILE   64
#define LDS_STR 80   // 64B of k-slice data + 16B pad (bank-conflict-free)

__global__ void __launch_bounds__(128)
apply_kernel(const _Float16* __restrict__ Kh, const _Float16* __restrict__ zt,
             const float* __restrict__ xin, float* __restrict__ xout) {
  __shared__ char lsa[2][MTILE * LDS_STR];  // 2 x 5 KB  (A: 64 rows x 32 k)
  __shared__ char lsb[2][ND * LDS_STR];     // 2 x 10 KB (B: 128 d  x 32 k)
  const int g    = blockIdx.y;
  const int tid  = threadIdx.x;
  const int lane = tid & 31;
  const int wave = tid >> 5;
  const int r0b  = blockIdx.x * MTILE;
  const int m = lane & 15, h = lane >> 4;

  const _Float16* Kg = Kh + (size_t)g * NP * NP;
  const _Float16* Zg = zt + (size_t)g * ND * NP;

  // Stage one 32-wide k-slice: A = 64x64B (256 xfers), B = 128x64B (512).
  // 128 threads -> 2 A-copies + 4 B-copies each (6 ASYNCcnt per iteration).
  auto stage = [&](int buf, int kb) {
    const char* KgB = (const char*)(Kg + (size_t)r0b * NP + kb);
    const char* ZgB = (const char*)(Zg + kb);
#pragma unroll
    for (int j = 0; j < 2; ++j) {
      const int idx = tid + j * 128, row = idx >> 2, q = idx & 3;
      copy16_to_lds(KgB + (size_t)row * (NP * 2) + q * 16,
                    &lsa[buf][row * LDS_STR + q * 16]);
    }
#pragma unroll
    for (int j = 0; j < 4; ++j) {
      const int idx = tid + j * 128, row = idx >> 2, q = idx & 3;
      copy16_to_lds(ZgB + (size_t)row * (NP * 2) + q * 16,
                    &lsb[buf][row * LDS_STR + q * 16]);
    }
  };

  stage(0, 0);
  v8f acc[8] = {};
  for (int it = 0; it < NP / 32; ++it) {
    const int buf = it & 1;
    if (it + 1 < NP / 32) {
      stage(buf ^ 1, (it + 1) * 32);  // prefetch next slice
      WAIT_ASYNC(6);                  // current slice (prev 6 xfers) landed
    } else {
      WAIT_ASYNC(0);
    }
    __syncthreads();
    // A fragment from LDS: row = wave*16 + m; chunks at h*16B and 32+h*16B.
    const char* arow = &lsa[buf][(wave * 16 + m) * LDS_STR];
    const v16h a = join16(*(const v8h*)(arow + h * 16),
                          *(const v8h*)(arow + 32 + h * 16));
#pragma unroll
    for (int nt = 0; nt < 8; ++nt) {
      const char* brow = &lsb[buf][(nt * 16 + m) * LDS_STR];
      const v16h b = join16(*(const v8h*)(brow + h * 16),
                            *(const v8h*)(brow + 32 + h * 16));
      acc[nt] = wmma_f16(a, b, acc[nt]);
    }
    __syncthreads();  // all waves done with buf before it is re-staged
  }

#pragma unroll
  for (int nt = 0; nt < 8; ++nt) {
    const int c = nt * 16 + m;
#pragma unroll
    for (int r = 0; r < 8; ++r) {
      const int rr = r0b + wave * 16 + 8 * h + r;
      const size_t idx = ((size_t)g * NP + rr) * ND + c;
      xout[idx] = 0.5f * acc[nt][r] + 0.5f * xin[idx];
    }
  }
}

// ---- 5) mean over rows -> output slot sIdx ---------------------------------
__global__ void mean_kernel(const float* __restrict__ x,
                            float* __restrict__ out, int sIdx) {
  const int g = blockIdx.x, d = threadIdx.x;
  const int b = g >> 2, w = g & 3;
  const float* xg = x + (size_t)g * NP * ND;
  float acc = 0.0f;
  for (int n = 0; n < NP; ++n) acc += xg[(size_t)n * ND + d];
  out[(size_t)b * (4 * 5 * ND) + w * (5 * ND) + sIdx * ND + d] =
      acc * (1.0f / (float)NP);
}

extern "C" void kernel_launch(void* const* d_in, const int* in_sizes, int n_in,
                              void* d_out, int out_size, void* d_ws,
                              size_t ws_size, hipStream_t stream) {
  (void)in_sizes; (void)n_in; (void)out_size; (void)ws_size;
  const float* pc = (const float*)d_in[0];  // [4,2048,128]
  const float* al = (const float*)d_in[1];  // [4,128]
  const float* sg = (const float*)d_in[2];  // scalar sigma
  float* out = (float*)d_out;               // [4, 2560]

  char* ws = (char*)d_ws;
  size_t off = 0;
  auto take = [&](size_t bytes) -> char* {
    char* p = ws + off;
    off += (bytes + 255) & ~(size_t)255;
    return p;
  };
  _Float16* Xh     = (_Float16*)take((size_t)NG * NP * ND * 2);   //  8.4 MB
  float*    rn     = (float*)take((size_t)NG * NP * 4);           //  128 KB
  float*    colsum = (float*)take((size_t)NG * NP * 4);           //  128 KB
  _Float16* Kh     = (_Float16*)take((size_t)NG * NP * NP * 2);   //  134 MB
  _Float16* zt     = (_Float16*)take((size_t)NG * ND * NP * 2);   //  8.4 MB
  float*    xA     = (float*)take((size_t)NG * NP * ND * 4);      // 16.8 MB
  float*    xB     = (float*)take((size_t)NG * NP * ND * 4);      // 16.8 MB

  xbar_kernel<<<dim3(NP, NG), ND, 0, stream>>>(pc, al, Xh, xA, rn, colsum);
  gram_kernel<<<dim3(NP / 64, NP / 128, NG), 256, 0, stream>>>(Xh, rn, colsum,
                                                               Kh, sg);
  mean_kernel<<<NG, ND, 0, stream>>>(xA, out, 0);

  float* cur = xA;
  float* nxt = xB;
  const int slot[9] = {0, 1, 2, 0, 3, 0, 0, 0, 4};
  for (int step = 1; step <= 8; ++step) {
    prescale_kernel<<<dim3(NP, NG), ND, 0, stream>>>(cur, colsum, zt);
    apply_kernel<<<dim3(NP / MTILE, NG), 128, 0, stream>>>(Kh, zt, cur, nxt);
    if (step == 1 || step == 2 || step == 4 || step == 8)
      mean_kernel<<<NG, ND, 0, stream>>>(nxt, out, slot[step]);
    float* t = cur; cur = nxt; nxt = t;
  }
}
